// STAN_21199958573597
// MI455X (gfx1250) — matrix operand order repaired and checked by reading, not compiled, not run on hardware
//
#include <hip/hip_runtime.h>
#include <cmath>

#define B_  64
#define L_  25
#define F_  128
#define P_  10000
#define BL_ (B_ * L_)

typedef __attribute__((ext_vector_type(2)))  float  v2f;
typedef __attribute__((ext_vector_type(8)))  float  v8f;
typedef __attribute__((ext_vector_type(8)))  __bf16 v8bf;
typedef __attribute__((ext_vector_type(16))) __bf16 v16bf;

__device__ inline v16bf cat16(v8bf a, v8bf b) {
    return __builtin_shufflevector(a, b, 0, 1, 2, 3, 4, 5, 6, 7,
                                   8, 9, 10, 11, 12, 13, 14, 15);
}
__device__ inline v8bf ld8(const __bf16* p) { return *(const v8bf*)p; }

// ---------- helpers: ordered-uint atomics (valid for non-negative floats) ----------
__device__ inline void atomicMaxNN(unsigned* p, float v) { atomicMax(p, __float_as_uint(v)); }
__device__ inline void atomicMinNN(unsigned* p, float v) { atomicMin(p, __float_as_uint(v)); }

// scal layout (uint bits of floats):
// [0]=ds_max [1]=ds_min [2]=dt_max [3]=dt_min [4]=s1_max [5]=s1_min [6]=t1_max [7]=t1_min
__global__ void k_init(unsigned* scal) {
    int i = threadIdx.x;
    if (i < 4) { scal[2 * i] = 0u; scal[2 * i + 1] = 0x7F800000u; }
}

// ---------- inputs = user_emb + poi_emb + tod_emb + dow_emb ----------
__global__ void k_embed(const int* __restrict__ user, const int* __restrict__ poi,
                        const int* __restrict__ tod, const int* __restrict__ dow,
                        const float* __restrict__ uw, const float* __restrict__ pw,
                        const float* __restrict__ tw, const float* __restrict__ dw,
                        float* __restrict__ inputs) {
    int bl = blockIdx.x;       // 0..1599
    int f  = threadIdx.x;      // 0..127
    inputs[bl * F_ + f] = uw[user[bl] * F_ + f] + pw[poi[bl] * F_ + f] +
                          tw[tod[bl] * F_ + f] + dw[dow[bl] * F_ + f];
}

// ---------- fp32 -> bf16 hi/lo split (x ~= hi + lo) ----------
__global__ void k_pack(const float* __restrict__ src, __bf16* __restrict__ hi,
                       __bf16* __restrict__ lo, int n) {
    int i = blockIdx.x * blockDim.x + threadIdx.x;
    if (i < n) {
        float x  = src[i];
        __bf16 h = (__bf16)x;
        hi[i] = h;
        lo[i] = (__bf16)(x - (float)h);
    }
}

// ---------- delta_s (haversine), delta_t, plus max/min reductions ----------
__global__ void k_delta_a(const float* __restrict__ lat, const float* __restrict__ lon,
                          const float* __restrict__ ut,
                          float* __restrict__ ds_out, float* __restrict__ dt_out,
                          unsigned* __restrict__ scal) {
    int idx = blockIdx.x * blockDim.x + threadIdx.x;
    float mx_s = 0.f, mn_s = INFINITY, mx_t = 0.f, mn_t = INFINITY;
    if (idx < B_ * L_ * L_) {
        int b = idx / (L_ * L_);
        int r = idx % (L_ * L_);
        int i = r / L_, j = r % L_;
        float la1 = lat[b * L_ + i], lo1 = lon[b * L_ + i];
        float la2 = lat[b * L_ + j], lo2 = lon[b * L_ + j];
        const float p = 3.14159265358979323846f / 180.0f;
        float a = 0.5f - __cosf((la2 - la1) * p) * 0.5f +
                  __cosf(la1 * p) * __cosf(la2 * p) * (1.0f - __cosf((lo2 - lo1) * p)) * 0.5f;
        float ds = 12742.0f * asinf(sqrtf(fmaxf(a, 0.0f)));
        float dt = fabsf(ut[b * L_ + i] - ut[b * L_ + j]);
        ds_out[idx] = ds; dt_out[idx] = dt;
        mx_s = mn_s = ds; mx_t = mn_t = dt;
    }
    #pragma unroll
    for (int o = 16; o > 0; o >>= 1) {
        mx_s = fmaxf(mx_s, __shfl_xor(mx_s, o, 32));
        mn_s = fminf(mn_s, __shfl_xor(mn_s, o, 32));
        mx_t = fmaxf(mx_t, __shfl_xor(mx_t, o, 32));
        mn_t = fminf(mn_t, __shfl_xor(mn_t, o, 32));
    }
    if ((threadIdx.x & 31) == 0) {
        atomicMaxNN(&scal[0], mx_s); atomicMinNN(&scal[1], mn_s);
        atomicMaxNN(&scal[2], mx_t); atomicMinNN(&scal[3], mn_t);
    }
}

// ---------- time_interval + its max/min ----------
__global__ void k_ti(const float* __restrict__ ut, float* __restrict__ ti,
                     unsigned* __restrict__ scal) {
    int idx = blockIdx.x * blockDim.x + threadIdx.x;
    float mx = 0.f, mn = INFINITY;
    if (idx < BL_) {
        int l = idx % L_;
        float v = (l == 0) ? 0.0f : fabsf(ut[idx] - ut[idx - 1]);
        ti[idx] = v;
        mx = mn = v;
    }
    #pragma unroll
    for (int o = 16; o > 0; o >>= 1) {
        mx = fmaxf(mx, __shfl_xor(mx, o, 32));
        mn = fminf(mn, __shfl_xor(mn, o, 32));
    }
    if ((threadIdx.x & 31) == 0) { atomicMaxNN(&scal[6], mx); atomicMinNN(&scal[7], mn); }
}

// ---------- max/min over distance_matrix[poi]: one block per gathered row ----------
__global__ void k_dmred(const int* __restrict__ poi, const float* __restrict__ dm,
                        unsigned* __restrict__ scal) {
    int bl = blockIdx.x;                        // 0..1599
    const float* row = dm + poi[bl] * P_;
    float mx = 0.f, mn = INFINITY;
    for (int p = threadIdx.x; p < P_; p += blockDim.x) {
        float v = row[p];
        mx = fmaxf(mx, v); mn = fminf(mn, v);
    }
    #pragma unroll
    for (int o = 16; o > 0; o >>= 1) {
        mx = fmaxf(mx, __shfl_xor(mx, o, 32));
        mn = fminf(mn, __shfl_xor(mn, o, 32));
    }
    if ((threadIdx.x & 31) == 0) { atomicMaxNN(&scal[4], mx); atomicMinNN(&scal[5], mn); }
}

// ---------- delta = (exp(-ds/(su-sl)) + exp(-dt/(tu-tl)))/2 ----------
__global__ void k_delta_b(const float* __restrict__ ds, const float* __restrict__ dt,
                          const unsigned* __restrict__ scal, float* __restrict__ delta) {
    int idx = blockIdx.x * blockDim.x + threadIdx.x;
    if (idx < B_ * L_ * L_) {
        float inv_s = 1.0f / (__uint_as_float(scal[0]) - __uint_as_float(scal[1]));
        float inv_t = 1.0f / (__uint_as_float(scal[2]) - __uint_as_float(scal[3]));
        delta[idx] = 0.5f * (__expf(-ds[idx] * inv_s) + __expf(-dt[idx] * inv_t));
    }
}

// ---------- QKV GEMM: (1600x128) @ (128x128) + bias, exact fp32 WMMA 16x16x4 ----------
__global__ void k_gemm_qkv(const float* __restrict__ inp, const float* __restrict__ W,
                           const float* __restrict__ bias, float* __restrict__ out) {
    int wave = (blockIdx.x * blockDim.x + threadIdx.x) >> 5;
    int lane = threadIdx.x & 31;
    const int NT = F_ / 16;                       // 8
    if (wave >= (BL_ / 16) * NT) return;          // wave-uniform -> EXEC stays all-1
    int mt = wave / NT, nt = wave % NT;
    int lane16 = lane & 15;
    int khalf  = (lane >> 4) << 1;                // 0 or 2
    int m = mt * 16 + lane16;
    int n = nt * 16 + lane16;
    v8f acc = {};
    #pragma unroll
    for (int k0 = 0; k0 < F_; k0 += 4) {
        int kk = k0 + khalf;
        v2f a, bf;
        a.x  = inp[m * F_ + kk];
        a.y  = inp[m * F_ + kk + 1];
        bf.x = W[kk * F_ + n];
        bf.y = W[(kk + 1) * F_ + n];
        acc = __builtin_amdgcn_wmma_f32_16x16x4_f32(false, a, false, bf,
                                                    (short)0, acc, false, false);
    }
    float bn = bias[n];
    int mhi = (lane >> 4) << 3;                   // 0 or 8
    #pragma unroll
    for (int r = 0; r < 8; r++) {
        out[(mt * 16 + r + mhi) * F_ + n] = acc[r] + bn;
    }
}

// ---------- scores + softmax + attn @ v (per-batch block) ----------
__global__ void k_attn(const float* __restrict__ q, const float* __restrict__ k,
                       const float* __restrict__ v, const float* __restrict__ delta,
                       float* __restrict__ attn_out) {
    int b = blockIdx.x;
    int t = threadIdx.x;                          // 256 threads
    __shared__ float sc[L_ * L_];
    __shared__ float w[L_ * L_];
    for (int e = t; e < L_ * L_; e += 256) {
        int i = e / L_, j = e % L_;
        const float* qi = q + (b * L_ + i) * F_;
        const float* kj = k + (b * L_ + j) * F_;
        float s = 0.f;
        for (int f = 0; f < F_; f++) s += qi[f] * kj[f];
        sc[e] = s + delta[b * L_ * L_ + e];
    }
    __syncthreads();
    if (t < L_) {
        float mx = -INFINITY;
        for (int j = 0; j < L_; j++) mx = fmaxf(mx, sc[t * L_ + j]);
        float sum = 0.f;
        for (int j = 0; j < L_; j++) { float e = __expf(sc[t * L_ + j] - mx); w[t * L_ + j] = e; sum += e; }
        float inv = 1.0f / sum;
        for (int j = 0; j < L_; j++) w[t * L_ + j] *= inv;
    }
    __syncthreads();
    for (int e = t; e < L_ * F_; e += 256) {
        int i = e / F_, f = e % F_;
        float s = 0.f;
        for (int j = 0; j < L_; j++) s += w[i * L_ + j] * v[(b * L_ + j) * F_ + f];
        attn_out[(b * L_ + i) * F_ + f] = s;
    }
}

// ---------- candidate GEMM (bf16x3 fp32-emulation) + interval epilogue ----------
// One wave per (b, p-tile of 16). D = poi_emb(16x128) @ attn_out[b]^T(128x25 -> 2 N-tiles)
// A/B split into bf16 hi+lo; D ~= Ah*Bh + Ah*Bl + Al*Bh accumulated in fp32.
__global__ void k_final(const __bf16* __restrict__ ph, const __bf16* __restrict__ pl,
                        const __bf16* __restrict__ aoh, const __bf16* __restrict__ aol,
                        const float* __restrict__ dm, const int* __restrict__ poi,
                        const float* __restrict__ ti, const unsigned* __restrict__ scal,
                        const float* __restrict__ w_val, const float* __restrict__ b_val,
                        float* __restrict__ out) {
    int wave = (blockIdx.x * blockDim.x + threadIdx.x) >> 5;
    int lane = threadIdx.x & 31;
    const int PT = P_ / 16;                       // 625
    if (wave >= B_ * PT) return;                  // wave-uniform
    int b  = wave / PT;
    int pt = wave % PT;
    int pbase  = pt * 16;
    int lane16 = lane & 15;
    int hi16   = lane >> 4;                       // 0 or 1
    int klo    = hi16 << 4;                       // A-frag K offset (0 or 16)
    int kbase  = hi16 << 4;                       // B-frag K offset (0 or 16)
    int mhi    = hi16 << 3;
    int mrow   = pbase + lane16;
    int n0 = lane16;                              // l = 0..15
    int n1 = 16 + lane16;                         // l = 16..31 (>=25 padded)
    bool ok1 = (n1 < L_);
    const __bf16* aohb = aoh + b * L_ * F_;
    const __bf16* aolb = aol + b * L_ * F_;
    v8f acc0 = {}, acc1 = {};
    #pragma unroll
    for (int kb = 0; kb < F_; kb += 32) {
        const __bf16* pA = ph + mrow * F_ + kb + klo;
        const __bf16* qA = pl + mrow * F_ + kb + klo;
        v16bf Ah = cat16(ld8(pA), ld8(pA + 8));
        v16bf Al = cat16(ld8(qA), ld8(qA + 8));
        const __bf16* pB0h = aohb + n0 * F_ + kb + kbase;
        const __bf16* pB0l = aolb + n0 * F_ + kb + kbase;
        v16bf B0h = cat16(ld8(pB0h), ld8(pB0h + 8));
        v16bf B0l = cat16(ld8(pB0l), ld8(pB0l + 8));
        v16bf B1h = {}, B1l = {};
        if (ok1) {
            const __bf16* pB1h = aohb + n1 * F_ + kb + kbase;
            const __bf16* pB1l = aolb + n1 * F_ + kb + kbase;
            B1h = cat16(ld8(pB1h), ld8(pB1h + 8));
            B1l = cat16(ld8(pB1l), ld8(pB1l + 8));
        }
        // correction terms first, main term last
        acc0 = __builtin_amdgcn_wmma_f32_16x16x32_bf16(false, Ah, false, B0l, (short)0, acc0, false, false);
        acc0 = __builtin_amdgcn_wmma_f32_16x16x32_bf16(false, Al, false, B0h, (short)0, acc0, false, false);
        acc0 = __builtin_amdgcn_wmma_f32_16x16x32_bf16(false, Ah, false, B0h, (short)0, acc0, false, false);
        acc1 = __builtin_amdgcn_wmma_f32_16x16x32_bf16(false, Ah, false, B1l, (short)0, acc1, false, false);
        acc1 = __builtin_amdgcn_wmma_f32_16x16x32_bf16(false, Al, false, B1h, (short)0, acc1, false, false);
        acc1 = __builtin_amdgcn_wmma_f32_16x16x32_bf16(false, Ah, false, B1h, (short)0, acc1, false, false);
    }
    float inv_s1 = 1.0f / (__uint_as_float(scal[4]) - __uint_as_float(scal[5]));
    float inv_t1 = 1.0f / (__uint_as_float(scal[6]) - __uint_as_float(scal[7]));
    float sum[8];
    #pragma unroll
    for (int r = 0; r < 8; r++) sum[r] = 0.0f;
    {   // N-tile 0: l = n0 (< 25 always)
        int l = n0;
        int dmrow = poi[b * L_ + l] * P_;
        float tiv = __expf(-ti[b * L_ + l] * inv_t1);
        float wv  = w_val[l];
        #pragma unroll
        for (int r = 0; r < 8; r++) {
            int p = pbase + r + mhi;
            float intv = 0.5f * (__expf(-dm[dmrow + p] * inv_s1) + tiv);
            sum[r] += acc0[r] * intv * wv;
        }
    }
    if (ok1) {  // N-tile 1: l = 16..24
        int l = n1;
        int dmrow = poi[b * L_ + l] * P_;
        float tiv = __expf(-ti[b * L_ + l] * inv_t1);
        float wv  = w_val[l];
        #pragma unroll
        for (int r = 0; r < 8; r++) {
            int p = pbase + r + mhi;
            float intv = 0.5f * (__expf(-dm[dmrow + p] * inv_s1) + tiv);
            sum[r] += acc1[r] * intv * wv;
        }
    }
    float bval = b_val[0];
    #pragma unroll
    for (int r = 0; r < 8; r++) {
        float s = sum[r];
        #pragma unroll
        for (int o = 1; o < 16; o <<= 1) s += __shfl_xor(s, o, 16);  // per-half reduce over l
        sum[r] = s;
    }
    if (lane16 == 0) {
        #pragma unroll
        for (int r = 0; r < 8; r++) {
            int p = pbase + r + mhi;
            float v = sum[r] + bval;
            out[b * P_ + p] = v;               // pre
            out[B_ * P_ + b * P_ + p] = v;     // second tuple element (pre, pre)
        }
    }
}

extern "C" void kernel_launch(void* const* d_in, const int* in_sizes, int n_in,
                              void* d_out, int out_size, void* d_ws, size_t ws_size,
                              hipStream_t stream) {
    const int*   user  = (const int*)d_in[0];
    const int*   poi   = (const int*)d_in[1];
    /* cat = d_in[2] unused by reference */
    const float* lat   = (const float*)d_in[3];
    const float* lon   = (const float*)d_in[4];
    const int*   tod   = (const int*)d_in[5];
    const int*   dow   = (const int*)d_in[6];
    const float* ut    = (const float*)d_in[7];
    const float* uw    = (const float*)d_in[8];
    const float* pw    = (const float*)d_in[9];
    const float* tw    = (const float*)d_in[10];
    const float* dw    = (const float*)d_in[11];
    const float* Wq    = (const float*)d_in[12];
    const float* bq    = (const float*)d_in[13];
    const float* Wk    = (const float*)d_in[14];
    const float* bk    = (const float*)d_in[15];
    const float* Wv    = (const float*)d_in[16];
    const float* bv    = (const float*)d_in[17];
    const float* w_val = (const float*)d_in[18];
    const float* b_val = (const float*)d_in[19];
    const float* dm    = (const float*)d_in[20];

    float* ws       = (float*)d_ws;
    float* inputs   = ws;                  // 204800
    float* q        = ws + 204800;         // 204800
    float* k        = ws + 409600;         // 204800
    float* v        = ws + 614400;         // 204800
    float* attn     = ws + 819200;         // 204800
    float* ds       = ws + 1024000;        // 40000
    float* dt       = ws + 1064000;        // 40000
    float* delta    = ws + 1104000;        // 40000
    float* ti       = ws + 1144000;        // 1600
    unsigned* scal  = (unsigned*)(ws + 1145600);   // 8 (+pad to 1145616)
    __bf16* poi_hi  = (__bf16*)(ws + 1145616);     // 1,280,000 bf16 = 640000 f
    __bf16* poi_lo  = (__bf16*)(ws + 1785616);     // 640000 f
    __bf16* ao_hi   = (__bf16*)(ws + 2425616);     // 204800 bf16 = 102400 f
    __bf16* ao_lo   = (__bf16*)(ws + 2528016);     // 102400 f  (total ~10.5 MB)

    k_init   <<<1, 32, 0, stream>>>(scal);
    k_embed  <<<BL_, F_, 0, stream>>>(user, poi, tod, dow, uw, pw, tw, dw, inputs);
    k_delta_a<<<(B_ * L_ * L_ + 255) / 256, 256, 0, stream>>>(lat, lon, ut, ds, dt, scal);
    k_ti     <<<(BL_ + 255) / 256, 256, 0, stream>>>(ut, ti, scal);
    k_dmred  <<<BL_, 256, 0, stream>>>(poi, dm, scal);
    k_delta_b<<<(B_ * L_ * L_ + 255) / 256, 256, 0, stream>>>(ds, dt, scal, delta);
    k_pack   <<<(P_ * F_ + 255) / 256, 256, 0, stream>>>(pw, poi_hi, poi_lo, P_ * F_);
    k_gemm_qkv<<<200, 128, 0, stream>>>(inputs, Wq, bq, q);
    k_gemm_qkv<<<200, 128, 0, stream>>>(inputs, Wk, bk, k);
    k_gemm_qkv<<<200, 128, 0, stream>>>(inputs, Wv, bv, v);
    k_attn   <<<B_, 256, 0, stream>>>(q, k, v, delta, attn);
    k_pack   <<<(BL_ * F_ + 255) / 256, 256, 0, stream>>>(attn, ao_hi, ao_lo, BL_ * F_);
    // 64 * 625 = 40000 waves, 8 waves per 256-thread block -> 5000 blocks
    k_final  <<<5000, 256, 0, stream>>>(poi_hi, poi_lo, ao_hi, ao_lo, dm, poi, ti, scal,
                                        w_val, b_val, (float*)d_out);
}
